// Conv2d_int8_35931696398712
// MI455X (gfx1250) — 8-bit matrix operand order repaired and checked by reading, compiled, not run on hardware
//
#include <hip/hip_runtime.h>

typedef int v8i __attribute__((ext_vector_type(8)));

#define BATCH 4
#define CIN   64
#define COUT  64
#define HH    28
#define WW    28
#define PH    30
#define NPIX  (HH*WW)
#define MT    (NPIX/16)
#define XQ_BYTES (BATCH*PH*PH*CIN)
#define WQ_BYTES (9*COUT*64)
#define N_X   (BATCH*CIN*HH*WW)
#define N_W   (COUT*CIN*9)

__global__ void absmax_kernel(const float* __restrict__ src, int n,
                              unsigned int* __restrict__ dst) {
    __shared__ float red[256];
    float m = 0.0f;
    for (int i = blockIdx.x * blockDim.x + threadIdx.x; i < n;
         i += gridDim.x * blockDim.x)
        m = fmaxf(m, fabsf(src[i]));
    red[threadIdx.x] = m;
    __syncthreads();
    for (int s = 128; s > 0; s >>= 1) {
        if (threadIdx.x < s)
            red[threadIdx.x] = fmaxf(red[threadIdx.x], red[threadIdx.x + s]);
        __syncthreads();
    }
    if (threadIdx.x == 0)
        atomicMax(dst, __float_as_uint(red[0]));
}

__device__ __forceinline__ signed char quant1(float v, float inv_s) {
    float r = rintf(v * inv_s);
    r = fminf(fmaxf(r, -128.0f), 127.0f);
    return (signed char)(int)r;
}

__global__ void quant_x_kernel(const float* __restrict__ x,
                               const unsigned int* __restrict__ scales,
                               signed char* __restrict__ xq) {
    int i = blockIdx.x * blockDim.x + threadIdx.x;
    if (i >= BATCH * PH * PH * CIN) return;
    int c  = i & 63;
    int t  = i >> 6;
    int xx = t % PH;  t /= PH;
    int y  = t % PH;
    int b  = t / PH;
    signed char q = 0;
    if (y >= 1 && y <= HH && xx >= 1 && xx <= WW) {
        float inv_s = 127.0f / __uint_as_float(scales[0]);
        float v = x[((b * CIN + c) * HH + (y - 1)) * WW + (xx - 1)];
        q = quant1(v, inv_s);
    }
    xq[i] = q;
}

__global__ void quant_w_kernel(const float* __restrict__ w,
                               const unsigned int* __restrict__ scales,
                               signed char* __restrict__ wq) {
    int i = blockIdx.x * blockDim.x + threadIdx.x;
    if (i >= 9 * COUT * 64) return;
    int c   = i & 63;
    int co  = (i >> 6) & 63;
    int tap = i >> 12;
    int kh = tap / 3, kw = tap % 3;
    float inv_s = 127.0f / __uint_as_float(scales[1]);
    float v = w[((co * CIN + c) * 3 + kh) * 3 + kw];
    wq[i] = quant1(v, inv_s);
}

__global__ __launch_bounds__(128)
void conv_wmma_kernel(const signed char* __restrict__ xq,
                      const signed char* __restrict__ wq,
                      const float* __restrict__ bias,
                      const unsigned int* __restrict__ scales,
                      float* __restrict__ out) {
    __shared__ __align__(16) signed char lds_w[WQ_BYTES];

    const int tid = threadIdx.x;

    {
        const unsigned long long gbase = (unsigned long long)(const void*)wq;
        #pragma unroll
        for (int i = 0; i < WQ_BYTES / 16 / 128; ++i) {
            unsigned int       lds_off = (unsigned int)(tid + i * 128) * 16u;
            unsigned long long gaddr   = gbase + (unsigned long long)(tid + i * 128) * 16ull;
            asm volatile("global_load_async_to_lds_b128 %0, %1, off"
                         :: "v"(lds_off), "v"(gaddr)
                         : "memory");
        }
        asm volatile("s_wait_asynccnt 0x0" ::: "memory");
    }
    __syncthreads();
    const signed char* ldsw = lds_w;

    const int mtile = blockIdx.x;
    const int b     = blockIdx.y;
    const int wave  = tid >> 5;
    const int lane  = tid & 31;
    const int mrow  = lane & 15;
    const int khalf = lane >> 4;

    const int pixel = mtile * 16 + mrow;
    const int oh = pixel / WW;
    const int ow = pixel % WW;

    v8i acc = {0, 0, 0, 0, 0, 0, 0, 0};

    #pragma unroll
    for (int tap = 0; tap < 9; ++tap) {
        const int kh = tap / 3, kw = tap % 3;

        const signed char* abase =
            xq + (((b * PH + (oh + kh)) * PH) + (ow + kw)) * 64 + khalf * 16;
        uint2 a0 = *(const uint2*)(abase + 0);
        uint2 a1 = *(const uint2*)(abase + 8);
        uint2 a2 = *(const uint2*)(abase + 32);
        uint2 a3 = *(const uint2*)(abase + 40);
        v8i A;
        A[0] = (int)a0.x; A[1] = (int)a0.y; A[2] = (int)a1.x; A[3] = (int)a1.y;
        A[4] = (int)a2.x; A[5] = (int)a2.y; A[6] = (int)a3.x; A[7] = (int)a3.y;

        const signed char* bbase =
            ldsw + tap * (COUT * 64) + (wave * 16 + mrow) * 64 + khalf * 16;
        int4 b0 = *(const int4*)(bbase + 0);
        int4 b1 = *(const int4*)(bbase + 32);
        v8i Bm;
        Bm[0] = b0.x; Bm[1] = b0.y; Bm[2] = b0.z; Bm[3] = b0.w;
        Bm[4] = b1.x; Bm[5] = b1.y; Bm[6] = b1.z; Bm[7] = b1.w;

        acc = __builtin_amdgcn_wmma_i32_16x16x64_iu8(true, A, true, Bm, acc,
                                                     false, false);
    }

    const float s = (__uint_as_float(scales[0]) * (1.0f / 127.0f)) *
                    (__uint_as_float(scales[1]) * (1.0f / 127.0f));
    const int co = wave * 16 + mrow;
    const float bv = bias[co];
    #pragma unroll
    for (int j = 0; j < 8; ++j) {
        int p = mtile * 16 + j + khalf * 8;
        out[(b * COUT + co) * NPIX + p] = (float)acc[j] * s + bv;
    }
}

extern "C" void kernel_launch(void* const* d_in, const int* in_sizes, int n_in,
                              void* d_out, int out_size, void* d_ws, size_t ws_size,
                              hipStream_t stream) {
    const float* x    = (const float*)d_in[0];
    const float* w    = (const float*)d_in[1];
    const float* bias = (const float*)d_in[2];
    float* out = (float*)d_out;

    unsigned char* ws     = (unsigned char*)d_ws;
    unsigned int*  scales = (unsigned int*)ws;
    signed char*   xq     = (signed char*)(ws + 1024);
    signed char*   wq     = (signed char*)(ws + 1024 + XQ_BYTES);

    hipMemsetAsync(d_ws, 0, 2 * sizeof(unsigned int), stream);

    absmax_kernel<<<256, 256, 0, stream>>>(x, N_X, scales + 0);
    absmax_kernel<<<72,  256, 0, stream>>>(w, N_W, scales + 1);

    quant_x_kernel<<<(BATCH * PH * PH * CIN + 255) / 256, 256, 0, stream>>>(x, scales, xq);
    quant_w_kernel<<<(9 * COUT * 64 + 255) / 256, 256, 0, stream>>>(w, scales, wq);

    conv_wmma_kernel<<<dim3(MT, BATCH), 128, 0, stream>>>(xq, wq, bias, scales, out);
}
